// HarmonicOscilattors_56659208569159
// MI455X (gfx1250) — matrix operand order repaired and checked by reading, compiled, not run on hardware
//
#include <hip/hip_runtime.h>
#include <hip/hip_bf16.h>

// ---------------------------------------------------------------------------
// HarmonicOscillators for MI455X (gfx1250, wave32, WMMA)
//
// Per 160-sample frame:  y = amp .* combine( S(160x128) x alpha16^T(128x16) )
// S[j,k] = antialias * sin_rev(frac((k+1)*P_j)),  phase kept mod 1 revolution
// (hardware v_sin_f32 consumes revolutions -> no 2pi scaling needed).
// GEMM tiled as 10 row-tiles x up-to-4 k-chunks of v_wmma_f32_16x16x32_f16,
// with wave-uniform antialias chunk skipping (mean live harmonics ~44/100).
// ---------------------------------------------------------------------------

#define SR        16000.0f
#define T_FRAMES  400
#define K_HARM    100
#define K_PAD     128
#define BLK       160          // upsample factor / samples per frame

typedef _Float16 v16h __attribute__((ext_vector_type(16)));
typedef float    v8f  __attribute__((ext_vector_type(8)));

// ---------------------------------------------------------------------------
// Kernel 1: per-batch fp64 exclusive scan of per-frame phase increments.
// Closed-form per-frame sum of upsampled f0/SR over its 160 samples:
//   sum_t = 20*F[max(t-1,0)] + 120*F[t] + 20*F[min(t+1,T-1)]
// Stores frac( C[t] - F[0] )  (phase in revolutions at frame start, with the
// reference's "minus first sample" correction folded in).
// ---------------------------------------------------------------------------
__global__ __launch_bounds__(512)
void ho_phase_scan(const float* __restrict__ f0, float* __restrict__ cadj) {
  const int b = blockIdx.x;
  const int t = threadIdx.x;           // 512 threads, 400 live
  __shared__ double sc[512];

  double fs = 0.0;
  if (t < T_FRAMES) {
    const float inv_sr = 1.0f / SR;
    float fm = f0[b * T_FRAMES + (t > 0 ? t - 1 : 0)] * inv_sr;
    float fc = f0[b * T_FRAMES + t] * inv_sr;
    float fp = f0[b * T_FRAMES + (t < T_FRAMES - 1 ? t + 1 : T_FRAMES - 1)] * inv_sr;
    fs = 20.0 * (double)fm + 120.0 * (double)fc + 20.0 * (double)fp;
  }
  sc[t] = fs;
  __syncthreads();

  // Hillis-Steele inclusive scan over 512 (zero padded)
  for (int off = 1; off < 512; off <<= 1) {
    double v = (t >= off) ? sc[t - off] : 0.0;
    __syncthreads();
    sc[t] += v;
    __syncthreads();
  }

  if (t < T_FRAMES) {
    double C  = (t == 0) ? 0.0 : sc[t - 1];        // exclusive prefix
    double F0 = (double)f0[b * T_FRAMES] * (1.0 / (double)SR);
    double ph = C - F0;                             // reference subtracts f0u[0]
    ph -= floor(ph);                                // keep mod 1 revolution
    cadj[b * T_FRAMES + t] = (float)ph;
  }
}

// ---------------------------------------------------------------------------
// Kernel 2: one block per (frame, batch). 320 threads = 10 wave32s.
// Wave w computes output samples [16w, 16w+16) via WMMA.
// ---------------------------------------------------------------------------
__global__ __launch_bounds__(320)
void ho_synth(const float* __restrict__ amp,
              const float* __restrict__ alpha,
              const float* __restrict__ f0,
              const float* __restrict__ cadj,
              float* __restrict__ out) {
  const int t   = blockIdx.x;     // frame
  const int b   = blockIdx.y;     // batch
  const int tid = threadIdx.x;

  __shared__ float    s_f0u[BLK];
  __shared__ float    s_amp[BLK];
  __shared__ float    s_P[BLK];
  __shared__ float    s_scan[256];
  __shared__ __align__(32) _Float16 s_alf[16][K_PAD];  // B tile: 3 live cols, rest 0
  __shared__ float    s_G[10][16][16];                 // per-tile D matrices

  const int r0 = (t > 0) ? t - 1 : 0;
  const int r1 = t;
  const int r2 = (t < T_FRAMES - 1) ? t + 1 : T_FRAMES - 1;

  // --- stage alpha rows (f32 -> f16); cols 3..15 and K 100..127 zeroed ---
  for (int i = tid; i < 16 * K_PAD; i += 320) {
    int col = i / K_PAD, kk = i % K_PAD;
    float v = 0.0f;
    if (col < 3 && kk < K_HARM) {
      int rr = (col == 0) ? r0 : ((col == 1) ? r1 : r2);
      v = alpha[((size_t)(b * T_FRAMES + rr)) * K_HARM + kk];
    }
    s_alf[col][kk] = (_Float16)v;
  }

  // --- per-sample upsampled controls (align_corners=False linear interp) ---
  if (tid < BLK) {
    float jc = (float)tid + 0.5f;
    float c0, c1, c2;
    if (tid < 80) { float w = 0.5f + jc * (1.0f / 160.0f); c0 = 1.0f - w; c1 = w; c2 = 0.0f; }
    else          { float w = jc * (1.0f / 160.0f) - 0.5f; c0 = 0.0f; c1 = 1.0f - w; c2 = w; }
    float fa = f0[b * T_FRAMES + r0], fb = f0[b * T_FRAMES + r1], fc = f0[b * T_FRAMES + r2];
    float aa = amp[b * T_FRAMES + r0], ab = amp[b * T_FRAMES + r1], ac = amp[b * T_FRAMES + r2];
    s_f0u[tid] = (c0 * fa + c1 * fb + c2 * fc) * (1.0f / SR);
    s_amp[tid] = c0 * aa + c1 * ab + c2 * ac;
  }
  __syncthreads();

  // --- within-frame inclusive scan of f0u (160 wide, fp32) ---
  if (tid < 256) s_scan[tid] = (tid < BLK) ? s_f0u[tid] : 0.0f;
  __syncthreads();
  for (int off = 1; off < 256; off <<= 1) {
    float v = 0.0f;
    if (tid < 256 && tid >= off) v = s_scan[tid - off];
    __syncthreads();
    if (tid < 256) s_scan[tid] += v;
    __syncthreads();
  }
  if (tid < BLK) {
    float P = cadj[b * T_FRAMES + t] + s_scan[tid];  // revolutions
    P -= floorf(P);
    s_P[tid] = P;
  }
  __syncthreads();

  // --- WMMA: each wave owns one 16-row tile; k-chunks of 32, vote to skip ---
  const int wave    = tid >> 5;
  const int lane    = tid & 31;
  const int hi_half = (lane >= 16) ? 1 : 0;
  const int row     = lane & 15;               // matrix row (A) / column (B)
  const int j       = wave * 16 + row;         // sample within frame

  const float P  = s_P[j];
  const float fu = s_f0u[j];
  // harmonic live iff  hk*fu < 0.5  and  hk <= 100  ->  hk < lim
  const float lim = fminf(0.5f / fu, (float)K_HARM + 0.5f);
  const float khalf = hi_half ? 16.0f : 0.0f;  // A-layout K offset for hi lanes

  v8f acc = {0.0f, 0.0f, 0.0f, 0.0f, 0.0f, 0.0f, 0.0f, 0.0f};

#pragma unroll
  for (int c = 0; c < 4; ++c) {
    const int kbase = c * 32;
    // chunk dead if even its smallest harmonic (kbase+1) is cut off for every
    // row of the tile -> wave-uniform skip of sins + WMMA (EXEC stays full).
    if (c == 0 || __any((float)(kbase + 1) < lim)) {
      // A operand: S tile synthesized in contiguous k order, matching B.
      // lane<16: elems 0..15 -> K=kbase+0..15
      // lane>=16: elems 0..15 -> K=kbase+16..31.
      const float kf = (float)(kbase + 1) + khalf;  // harmonic of element 0
      v16h a;
#pragma unroll
      for (int i = 0; i < 16; ++i) {
        float hk = kf + (float)i;
        float x  = __builtin_amdgcn_fractf(hk * P);     // phase in revolutions
        float sv = __builtin_amdgcn_sinf(x);            // v_sin_f32: sin(2*pi*x)
        a[i] = (_Float16)((hk < lim) ? sv : 0.0f);
      }

      // B operand: straight 32B load from the zero-padded column-major tile.
      // lane<16: K=kbase+0..15 ; lane>=16: K=kbase+16..31 ; column = lane&15.
      const v16h bb = *(const v16h*)&s_alf[row][kbase + (hi_half ? 16 : 0)];

      acc = __builtin_amdgcn_wmma_f32_16x16x32_f16(
          /*neg_a=*/false, a, /*neg_b=*/false, bb,
          /*c_mod=*/(short)0, acc, /*reuse_a=*/false, /*reuse_b=*/false);
    }
  }

  // D layout: vgpr r -> (M = r + 8*hi_half, N = lane&15)
#pragma unroll
  for (int r = 0; r < 8; ++r) {
    s_G[wave][r + (hi_half ? 8 : 0)][row] = acc[r];
  }
  __syncthreads();

  // --- combine the 3 dot products with interp weights, scale by amp ---
  if (tid < BLK) {
    int tile = tid >> 4, m = tid & 15;
    float jc = (float)tid + 0.5f;
    float c0, c1, c2;
    if (tid < 80) { float w = 0.5f + jc * (1.0f / 160.0f); c0 = 1.0f - w; c1 = w; c2 = 0.0f; }
    else          { float w = jc * (1.0f / 160.0f) - 0.5f; c0 = 0.0f; c1 = 1.0f - w; c2 = w; }
    float y = s_amp[tid] *
              (c0 * s_G[tile][m][0] + c1 * s_G[tile][m][1] + c2 * s_G[tile][m][2]);
    out[(size_t)b * (T_FRAMES * BLK) + (size_t)t * BLK + tid] = y;
  }
}

// ---------------------------------------------------------------------------
extern "C" void kernel_launch(void* const* d_in, const int* in_sizes, int n_in,
                              void* d_out, int out_size, void* d_ws, size_t ws_size,
                              hipStream_t stream) {
  const float* amp   = (const float*)d_in[0];   // [B,T,1]
  const float* alpha = (const float*)d_in[1];   // [B,T,K]
  const float* f0    = (const float*)d_in[2];   // [B,T,1]
  // d_in[3] = k (1..K), implicit in the kernel
  float* out  = (float*)d_out;
  float* cadj = (float*)d_ws;                   // [B,T] frame phase offsets (12.8 KB)

  const int B = in_sizes[2] / T_FRAMES;

  ho_phase_scan<<<dim3(B), dim3(512), 0, stream>>>(f0, cadj);
  ho_synth<<<dim3(T_FRAMES, B), dim3(320), 0, stream>>>(amp, alpha, f0, cadj, out);
}
